// Attention_18399639896382
// MI455X (gfx1250) — compile-verified, matrix operand order checked by reading, not run on hardware
//
#include <hip/hip_runtime.h>

typedef __attribute__((ext_vector_type(16))) _Float16 v16h;
typedef __attribute__((ext_vector_type(8)))  float    v8f;

#define B_      32
#define L_      4096
#define D_      256
#define CHUNKS  16          // L_ / 256
#define NEG_BIG 1e30f

// Load 8 f32 at p[0..7] and 8 f32 at p[16..23], convert to f16, pack into v16h.
// This matches the per-lane K ordering of the 16-bit WMMA A/B operand layout.
__device__ __forceinline__ void load_cvt16(v16h& v, const float* __restrict__ p) {
    const float4* q = (const float4*)p;
    float4 x0 = q[0], x1 = q[1];   // elements 0..7
    float4 x2 = q[4], x3 = q[5];   // elements 16..23
    v[0]  = (_Float16)x0.x; v[1]  = (_Float16)x0.y; v[2]  = (_Float16)x0.z; v[3]  = (_Float16)x0.w;
    v[4]  = (_Float16)x1.x; v[5]  = (_Float16)x1.y; v[6]  = (_Float16)x1.z; v[7]  = (_Float16)x1.w;
    v[8]  = (_Float16)x2.x; v[9]  = (_Float16)x2.y; v[10] = (_Float16)x2.z; v[11] = (_Float16)x2.w;
    v[12] = (_Float16)x3.x; v[13] = (_Float16)x3.y; v[14] = (_Float16)x3.z; v[15] = (_Float16)x3.w;
}

// Pass 1: scores[r] = dot(Q[r,:], kq) for r in [0, B*L). One wave -> 16 rows via
// 8x v_wmma_f32_16x16x32_f16 (B operand = kq broadcast across all 16 columns).
__global__ void k1_scores(const float* __restrict__ Q,
                          const float* __restrict__ kern,   // kq = kern[0:256]
                          float* __restrict__ scores) {
    const int lane  = threadIdx.x & 31;
    const int wave  = threadIdx.x >> 5;
    const long long row0 = (long long)blockIdx.x * 128 + (long long)wave * 16;
    const int m     = lane & 15;            // matrix row owned by this lane
    const int khalf = (lane >> 4) * 8;      // lanes 16-31 own K offset +8 per ISA layout

    const float* qrow = Q + (row0 + m) * D_;

    v8f c = {};
#pragma unroll
    for (int d0 = 0; d0 < D_; d0 += 32) {
        const int bd = d0 + khalf;
        v16h a, b;
        load_cvt16(a, qrow + bd);           // A: Q tile (16x32, f16)
        load_cvt16(b, kern + bd);           // B: kq chunk replicated over all columns
        c = __builtin_amdgcn_wmma_f32_16x16x32_f16(
                /*neg_a=*/false, a, /*neg_b=*/false, b,
                /*c_mod=*/(short)0, c, /*reuse_a=*/false, /*reuse_b=*/false);
    }

    // D layout: VGPR i holds M=i (lanes 0-15) and M=8+i (lanes 16-31); all N equal.
    if (m == 0) {
        const long long base = row0 + (long long)(lane >> 4) * 8;
#pragma unroll
        for (int i = 0; i < 8; i++) scores[base + i] = c[i];
    }
}

// Pass 2: per-batch masked softmax over L. One 256-thread block per batch.
// Also adds the per-batch W.kw + bias term (shift-invariant, kept for fidelity).
__global__ void k2_softmax(const float* __restrict__ W,
                           const unsigned char* __restrict__ mask,
                           const float* __restrict__ kern,   // kw = kern[256:512]
                           const float* __restrict__ bias,
                           float* __restrict__ scores) {     // in: scores, out: alpha
    const int b = blockIdx.x;
    const int t = threadIdx.x;
    __shared__ float red[256];

    // wb = dot(W[b,:], kw) + bias
    red[t] = W[b * D_ + t] * kern[D_ + t];
    __syncthreads();
    for (int s = 128; s > 0; s >>= 1) { if (t < s) red[t] += red[t + s]; __syncthreads(); }
    const float wb = red[0] + bias[0];
    __syncthreads();

    float sv[CHUNKS];
    float mx = -3.0e38f;
#pragma unroll
    for (int i = 0; i < CHUNKS; i++) {
        const int l = i * 256 + t;
        const float mm = mask[b * L_ + l] ? 1.0f : 0.0f;
        const float s  = scores[b * L_ + l] + wb - (1.0f - mm) * NEG_BIG;
        sv[i] = s;
        mx = fmaxf(mx, s);
    }
    red[t] = mx; __syncthreads();
    for (int s = 128; s > 0; s >>= 1) { if (t < s) red[t] = fmaxf(red[t], red[t + s]); __syncthreads(); }
    const float M = red[0];
    __syncthreads();

    float sum = 0.0f;
#pragma unroll
    for (int i = 0; i < CHUNKS; i++) { sv[i] = __expf(sv[i] - M); sum += sv[i]; }
    red[t] = sum; __syncthreads();
    for (int s = 128; s > 0; s >>= 1) { if (t < s) red[t] += red[t + s]; __syncthreads(); }
    const float inv = 1.0f / red[0];

#pragma unroll
    for (int i = 0; i < CHUNKS; i++) scores[b * L_ + i * 256 + t] = sv[i] * inv;
}

// Pass 3: partial[b, c, d] = sum over 256 l's in chunk c of alpha[b,l]*Q[b,l,d].
// Thread t owns output column d=t -> fully coalesced 1KB row loads from Q (L2-resident).
__global__ void k3_partial(const float* __restrict__ Q,
                           const float* __restrict__ alpha,
                           float* __restrict__ partial) {
    const int b = blockIdx.x;
    const int c = blockIdx.y;
    const int t = threadIdx.x;
    __shared__ float as[256];
    as[t] = alpha[b * L_ + c * 256 + t];
    __syncthreads();

    const float* qp = Q + ((long long)b * L_ + (long long)c * 256) * D_ + t;
    float acc = 0.0f;
#pragma unroll 8
    for (int l = 0; l < 256; l++)
        acc = fmaf(as[l], qp[(long long)l * D_], acc);

    partial[((long long)b * CHUNKS + c) * D_ + t] = acc;
}

// Pass 4: deterministic reduction of the 16 chunk partials into d_out.
__global__ void k4_reduce(const float* __restrict__ partial, float* __restrict__ out) {
    const int b = blockIdx.x;
    const int t = threadIdx.x;
    float s = 0.0f;
#pragma unroll
    for (int c = 0; c < CHUNKS; c++) s += partial[((long long)b * CHUNKS + c) * D_ + t];
    out[b * D_ + t] = s;
}

extern "C" void kernel_launch(void* const* d_in, const int* in_sizes, int n_in,
                              void* d_out, int out_size, void* d_ws, size_t ws_size,
                              hipStream_t stream) {
    const float*         Q    = (const float*)d_in[0];
    const float*         W    = (const float*)d_in[1];
    const unsigned char* mask = (const unsigned char*)d_in[2];
    const float*         kern = (const float*)d_in[3];
    const float*         bias = (const float*)d_in[4];
    float*               out  = (float*)d_out;

    float* scores  = (float*)d_ws;            // B*L floats (reused as alpha)
    float* partial = scores + B_ * L_;        // B*CHUNKS*D floats

    k1_scores <<<dim3((B_ * L_) / 128), dim3(256), 0, stream>>>(Q, kern, scores);
    k2_softmax<<<dim3(B_),              dim3(256), 0, stream>>>(W, mask, kern, bias, scores);
    k3_partial<<<dim3(B_, CHUNKS),      dim3(256), 0, stream>>>(Q, scores, partial);
    k4_reduce <<<dim3(B_),              dim3(256), 0, stream>>>(partial, out);
}
